// RealMetricsFastSpeech2_80728205295985
// MI455X (gfx1250) — compile-verified, matrix operand order checked by reading, not run on hardware
//
#include <hip/hip_runtime.h>

// ---------------- types / helpers ----------------
typedef __bf16 bf16;
typedef bf16  v16bf __attribute__((ext_vector_type(16)));
typedef bf16  v8bf  __attribute__((ext_vector_type(8)));
typedef float v8f   __attribute__((ext_vector_type(8)));
typedef float v4f   __attribute__((ext_vector_type(4)));

// Native f32 -> bf16 convert (RNE on gfx1250); far cheaper than manual bit-twiddled RNE.
__device__ __forceinline__ bf16 f2bf(float f) { return (bf16)f; }

// Fragment k-map (ISA 7.12.2, bf16 16x32): element e 0..7 -> k0+koff+e (contiguous),
// e 8..15 -> k0+16+koff+(e-8) (contiguous). koff = 0 (lanes 0-15) / 8 (lanes 16-31).
// => a fragment is four b128 loads (f32 source) or two b128 loads (bf16 source).
__device__ __forceinline__ v16bf load_frag_f32(const float* __restrict__ p) {
  v4f x0 = *(const v4f*)(p + 0);
  v4f x1 = *(const v4f*)(p + 4);
  v4f x2 = *(const v4f*)(p + 16);
  v4f x3 = *(const v4f*)(p + 20);
  v16bf v;
#pragma unroll
  for (int i = 0; i < 4; ++i) {
    v[i]      = f2bf(x0[i]);
    v[4 + i]  = f2bf(x1[i]);
    v[8 + i]  = f2bf(x2[i]);
    v[12 + i] = f2bf(x3[i]);
  }
  return v;
}

__device__ __forceinline__ v16bf load_frag_bf16(const bf16* __restrict__ p) {
  v8bf a = *(const v8bf*)(p);
  v8bf b = *(const v8bf*)(p + 16);
  return __builtin_shufflevector(a, b, 0,1,2,3,4,5,6,7,8,9,10,11,12,13,14,15);
}

// 8 contiguous f32 (one run), scaled by 0/1 for conv time-padding; load is unconditional
// (address pre-clamped by caller) so no exec-mask branching is generated.
__device__ __forceinline__ void load8_f32_masked(v16bf& v, int base,
                                                 const float* __restrict__ p, float s) {
  v4f x0 = *(const v4f*)(p);
  v4f x1 = *(const v4f*)(p + 4);
#pragma unroll
  for (int i = 0; i < 4; ++i) {
    v[base + i]     = f2bf(x0[i] * s);
    v[base + 4 + i] = f2bf(x1[i] * s);
  }
}

#define WMMA_BF16(a, b, c) \
  __builtin_amdgcn_wmma_f32_16x16x32_bf16(false, (a), false, (b), (short)0, (c), false, false)

// ---------------- constants ----------------
#define BB    32
#define SS    128
#define HH    256
#define H3    768
#define H4    1024
#define DH    128
#define NHH   2
#define TMEL  768
#define NMELC 80

// ---------------- linear: C[M,N] = act((A @ Wb^T) + bias); Wb bf16 [N,K] ----------------
// Requires M%16==0, N%16==0, K%32==0 (true for every call site).
__global__ __launch_bounds__(32)
void k_linear(const float* __restrict__ A, const bf16* __restrict__ Wb,
              const float* __restrict__ bias, float* __restrict__ C,
              int M, int N, int K, int act, float lo, float hi) {
  int tn = blockIdx.x, tm = blockIdx.y;
  int lane = threadIdx.x & 31;
  int row  = lane & 15;
  int hi16 = lane >> 4;
  int koff = hi16 << 3;
  int m = tm * 16 + row;
  int n = tn * 16 + row;
  const float* arow = A  + (long)m * K + koff;
  const bf16*  wrow = Wb + (long)n * K + koff;
  v8f acc = {};
#pragma unroll 2
  for (int k0 = 0; k0 < K; k0 += 32) {
    v16bf va = load_frag_f32(arow + k0);
    v16bf vb = load_frag_bf16(wrow + k0);
    acc = WMMA_BF16(va, vb, acc);
  }
  float bv = bias ? bias[n] : 0.f;
#pragma unroll
  for (int r = 0; r < 8; ++r) {
    int cm = tm * 16 + r + (hi16 ? 8 : 0);
    float v = acc[r] + bv;
    if (act == 1) v = fmaxf(v, 0.f);
    else if (act == 2) v = fminf(fmaxf(v, lo), hi);
    C[(long)cm * N + n] = v;
  }
}

// ---------------- attention scores: S[z,q,k] = (Q.K)/sqrt(dh), z = b*NH+h ----------------
__global__ __launch_bounds__(32)
void k_attn_scores(const float* __restrict__ QKV, float* __restrict__ S, int T, float scale) {
  int z = blockIdx.z, b = z >> 1, h = z & 1;
  int tn = blockIdx.x, tm = blockIdx.y;
  int lane = threadIdx.x & 31;
  int row = lane & 15, hi16 = lane >> 4, koff = hi16 << 3;
  int m = tm * 16 + row, n = tn * 16 + row;
  const float* qb = QKV + (long)b * T * H3 + h * DH;
  const float* arow = qb + (long)m * H3 + koff;            // Q row
  const float* brow = qb + HH + (long)n * H3 + koff;       // K row (+H channels)
  v8f acc = {};
#pragma unroll
  for (int k0 = 0; k0 < DH; k0 += 32) {
    acc = WMMA_BF16(load_frag_f32(arow + k0), load_frag_f32(brow + k0), acc);
  }
  float* sc = S + (long)z * T * T;
#pragma unroll
  for (int r = 0; r < 8; ++r) {
    int cm = tm * 16 + r + (hi16 ? 8 : 0);
    sc[(long)cm * T + n] = acc[r] * scale;
  }
}

// ---------------- masked softmax (one wave per row) ----------------
__global__ __launch_bounds__(32)
void k_softmax(float* __restrict__ S, const int* __restrict__ lenArr, int T) {
  int rid = blockIdx.x;                 // z*T + q
  int z = rid / T, b = z >> 1;
  float* p = S + (long)rid * T;
  int len = lenArr[b];
  len = len < 1 ? 1 : (len > T ? T : len);
  int lane = threadIdx.x & 31;
  float mx = -3.4e38f;
  for (int j = lane; j < len; j += 32) mx = fmaxf(mx, p[j]);
  for (int o = 16; o; o >>= 1) mx = fmaxf(mx, __shfl_xor(mx, o, 32));
  float sum = 0.f;
  for (int j = lane; j < len; j += 32) sum += __expf(p[j] - mx);
  for (int o = 16; o; o >>= 1) sum += __shfl_xor(sum, o, 32);
  float inv = 1.f / sum;
  for (int j = lane; j < T; j += 32) p[j] = (j < len) ? __expf(p[j] - mx) * inv : 0.f;
}

// ---------------- V transpose+cvt: VT[z][d][k] = bf16(V[b][k][h*dh+d]) ----------------
__global__ void k_vtrans(const float* __restrict__ QKV, bf16* __restrict__ VT, int T, int total) {
  int i = blockIdx.x * blockDim.x + threadIdx.x;
  if (i >= total) return;
  int k = i % T;
  int n = (i / T) % DH;
  int z = i / (T * DH);
  int b = z >> 1, h = z & 1;
  VT[i] = f2bf(QKV[((long)b * T + k) * H3 + 2 * HH + h * DH + n]);
}

// ---------------- O[b,q,h*dh+d] = sum_k P[z,q,k] * V[b,k,h*dh+d] ----------------
__global__ __launch_bounds__(32)
void k_attn_av(const float* __restrict__ P, const bf16* __restrict__ VT,
               float* __restrict__ O, int T) {
  int z = blockIdx.z, b = z >> 1, h = z & 1;
  int tn = blockIdx.x, tm = blockIdx.y;
  int lane = threadIdx.x & 31;
  int row = lane & 15, hi16 = lane >> 4, koff = hi16 << 3;
  int m = tm * 16 + row, n = tn * 16 + row;            // n < 128
  const float* arow = P + (long)z * T * T + (long)m * T + koff;
  const bf16*  brow = VT + ((long)z * DH + n) * T + koff;
  v8f acc = {};
#pragma unroll 2
  for (int k0 = 0; k0 < T; k0 += 32) {
    acc = WMMA_BF16(load_frag_f32(arow + k0), load_frag_bf16(brow + k0), acc);
  }
  float* ob = O + (long)b * T * HH + h * DH;
#pragma unroll
  for (int r = 0; r < 8; ++r) {
    int cm = tm * 16 + r + (hi16 ? 8 : 0);
    ob[(long)cm * HH + n] = acc[r];
  }
}

// ---------------- conv1d as implicit-im2col GEMM; Wt bf16 [Cout][dk*Cin+cin] ----------------
// k-order = dk*Cin+cin; an 8-run never crosses a Cin boundary (Cin % 32 == 0).
__global__ __launch_bounds__(32)
void k_conv_gemm(const float* __restrict__ X, const bf16* __restrict__ Wt,
                 const float* __restrict__ bias, float* __restrict__ Y,
                 int T, int Cin, int cinShift, int Cout, int pad, int KW, int relu) {
  int z = blockIdx.z;
  const float* xb = X + (long)z * T * Cin;
  int Kdim = Cin * KW;
  int tn = blockIdx.x, tm = blockIdx.y;
  int lane = threadIdx.x & 31;
  int row = lane & 15, hi16 = lane >> 4, koff = hi16 << 3;
  int m = tm * 16 + row, n = tn * 16 + row;
  const bf16* wrow = Wt + (long)n * Kdim + koff;
  v8f acc = {};
  for (int k0 = 0; k0 < Kdim; k0 += 32) {
    int r0 = k0 + koff, r1 = r0 + 16;
    int dk0 = r0 >> cinShift, ci0 = r0 & (Cin - 1);
    int dk1 = r1 >> cinShift, ci1 = r1 & (Cin - 1);
    int t0 = m + dk0 - pad,   t1 = m + dk1 - pad;
    float s0 = (t0 >= 0 && t0 < T) ? 1.f : 0.f;
    float s1 = (t1 >= 0 && t1 < T) ? 1.f : 0.f;
    int t0c = (t0 >= 0 && t0 < T) ? t0 : 0;
    int t1c = (t1 >= 0 && t1 < T) ? t1 : 0;
    v16bf va;
    load8_f32_masked(va, 0, xb + (long)t0c * Cin + ci0, s0);
    load8_f32_masked(va, 8, xb + (long)t1c * Cin + ci1, s1);
    v16bf vb = load_frag_bf16(wrow + k0);
    acc = WMMA_BF16(va, vb, acc);
  }
  float bv = bias[n];
  float* yb = Y + (long)z * T * Cout;
#pragma unroll
  for (int r = 0; r < 8; ++r) {
    int cm = tm * 16 + r + (hi16 ? 8 : 0);
    float v = acc[r] + bv;
    if (relu) v = fmaxf(v, 0.f);
    yb[(long)cm * Cout + n] = v;
  }
}

// ---------------- weight pre-transforms ----------------
__global__ void k_w2bf(const float* __restrict__ W, bf16* __restrict__ Wb, int n) {
  int i = blockIdx.x * blockDim.x + threadIdx.x;
  if (i < n) Wb[i] = f2bf(W[i]);
}
// Wt[n][dk*Cin+cin] = W[(n*Cin+cin)*KW+dk]
__global__ void k_wconvt(const float* __restrict__ W, bf16* __restrict__ Wt,
                         int Cin, int Cout, int KW) {
  int i = blockIdx.x * blockDim.x + threadIdx.x;
  int tot = Cout * Cin * KW;
  if (i >= tot) return;
  int n = i / (Cin * KW);
  int r = i - n * (Cin * KW);
  int dk = r / Cin;
  int ci = r - dk * Cin;
  Wt[i] = f2bf(W[((long)n * Cin + ci) * KW + dk]);
}

// ---------------- fused residual + LayerNorm over H=256 (one wave per row) ----------------
__global__ __launch_bounds__(32)
void k_add_ln(const float* __restrict__ X, const float* __restrict__ R,
              const float* __restrict__ g, const float* __restrict__ bb,
              float* __restrict__ Y) {
  long base = (long)blockIdx.x * HH;
  int lane = threadIdx.x & 31;
  float v[8];
  float s = 0.f, s2 = 0.f;
#pragma unroll
  for (int i = 0; i < 8; ++i) {
    int h = lane + i * 32;
    float t = X[base + h] + (R ? R[base + h] : 0.f);
    v[i] = t; s += t; s2 += t * t;
  }
  for (int o = 16; o; o >>= 1) { s += __shfl_xor(s, o, 32); s2 += __shfl_xor(s2, o, 32); }
  float mean = s * (1.f / HH);
  float var  = s2 * (1.f / HH) - mean * mean;
  float rstd = rsqrtf(var + 1e-5f);
#pragma unroll
  for (int i = 0; i < 8; ++i) {
    int h = lane + i * 32;
    Y[base + h] = (v[i] - mean) * rstd * g[h] + bb[h];
  }
}

// ---------------- embedding gather ----------------
__global__ void k_embed(const int* __restrict__ ids, const float* __restrict__ emb,
                        float* __restrict__ X) {
  int bt = blockIdx.x, h = threadIdx.x;
  X[(long)bt * HH + h] = emb[(long)ids[bt] * HH + h];
}

// ---------------- length regulation: cumsum + searchsorted ----------------
__global__ void k_lenreg(const int* __restrict__ dur, int* __restrict__ tok,
                         int* __restrict__ melLen, float* __restrict__ melLenOut) {
  __shared__ int cs[SS];
  __shared__ int totalS;
  int b = blockIdx.x;
  if (threadIdx.x == 0) {
    int acc = 0;
    for (int i = 0; i < SS; ++i) { acc += dur[b * SS + i]; cs[i] = acc; }
    totalS = acc;
    int ml = acc < 1 ? 1 : acc;
    melLen[b] = ml;
    melLenOut[b] = (float)ml;
  }
  __syncthreads();
  int total = totalS;
  for (int pos = threadIdx.x; pos < TMEL; pos += blockDim.x) {
    int lo = 0, hb = SS;
    while (lo < hb) { int mid = (lo + hb) >> 1; if (cs[mid] <= pos) lo = mid + 1; else hb = mid; }
    int t = lo < SS - 1 ? lo : SS - 1;
    tok[b * TMEL + pos] = (pos < total) ? t : -1;
  }
}

// ---------------- expand + clip one scalar feature along regulated time ----------------
__global__ void k_expand(const float* __restrict__ feat, const int* __restrict__ tok,
                         float* __restrict__ ef, int n, float lo, float hi) {
  int i = blockIdx.x * blockDim.x + threadIdx.x;
  if (i >= n) return;
  int b = i / TMEL;
  int t = tok[i];
  float v = (t >= 0) ? feat[b * SS + t] : 0.f;
  ef[i] = fminf(fmaxf(v, lo), hi);
}

// ---------------- build decoder input: gather + five 1-channel k=3 variance convs ----------------
struct VE5 { const float* w[5]; const float* b[5]; };

__global__ void k_build_dec(const float* __restrict__ Xenc, const int* __restrict__ tok,
                            const float* __restrict__ EF, VE5 ve, float* __restrict__ dec) {
  int bt = blockIdx.x;
  int b = bt / TMEL, t = bt - b * TMEL;
  int h = threadIdx.x;
  int tk = tok[bt];
  float acc = (tk >= 0) ? Xenc[((long)b * SS + tk) * HH + h] : 0.f;
#pragma unroll
  for (int f = 0; f < 5; ++f) {
    const float* efb = EF + (long)f * BB * TMEL + (long)b * TMEL;
    float xm = (t > 0) ? efb[t - 1] : 0.f;
    float xc = efb[t];
    float xp = (t < TMEL - 1) ? efb[t + 1] : 0.f;
    const float* wv = ve.w[f];
    acc += wv[h * 3 + 0] * xm + wv[h * 3 + 1] * xc + wv[h * 3 + 2] * xp + ve.b[f][h];
  }
  dec[(long)bt * HH + h] = acc;
}

// ---------------- variance prediction = clip(ob, lo, hi) broadcast ----------------
__global__ void k_fill_clip(const float* __restrict__ obp, float* __restrict__ out,
                            int n, float lo, float hi) {
  int i = blockIdx.x * blockDim.x + threadIdx.x;
  if (i < n) out[i] = fminf(fmaxf(obp[0], lo), hi);
}

// ================= host side =================
struct FFTP {
  const float *wqkv, *bqkv, *wo, *bo, *g1, *b1, *cw1, *cb1, *cw2, *cb2, *g2, *b2;
};
struct WsP {
  float *X, *X1, *O, *TMP, *QKV, *SC;  // SC doubles as FFN hidden C1
  float *XENC, *EF;
  int *TOK, *MLEN;
  bf16 *WT1, *WT2, *WQB, *WOB, *WMEL, *VT;
};

static void run_fft_block(const FFTP& p, float* X, int T, const int* lenArr,
                          const WsP& w, hipStream_t s) {
  const int M = BB * T;
  dim3 wv(32);
  // --- attention ---
  k_w2bf<<<dim3((H3 * HH + 255) / 256), dim3(256), 0, s>>>(p.wqkv, w.WQB, H3 * HH);
  k_linear<<<dim3(H3 / 16, M / 16), wv, 0, s>>>(X, w.WQB, p.bqkv, w.QKV, M, H3, HH, 0, 0.f, 0.f);
  k_attn_scores<<<dim3(T / 16, T / 16, BB * NHH), wv, 0, s>>>(w.QKV, w.SC, T, 0.08838834764831845f);
  k_softmax<<<dim3(BB * NHH * T), wv, 0, s>>>(w.SC, lenArr, T);
  int vtot = BB * NHH * DH * T;
  k_vtrans<<<dim3((vtot + 255) / 256), dim3(256), 0, s>>>(w.QKV, w.VT, T, vtot);
  k_attn_av<<<dim3(DH / 16, T / 16, BB * NHH), wv, 0, s>>>(w.SC, w.VT, w.O, T);
  k_w2bf<<<dim3((HH * HH + 255) / 256), dim3(256), 0, s>>>(p.wo, w.WOB, HH * HH);
  k_linear<<<dim3(HH / 16, M / 16), wv, 0, s>>>(w.O, w.WOB, p.bo, w.TMP, M, HH, HH, 0, 0.f, 0.f);
  k_add_ln<<<dim3(M), wv, 0, s>>>(X, w.TMP, p.g1, p.b1, w.X1);
  // --- conv FFN (k=9): 256 -> 1024 (relu) -> 256 ---
  k_wconvt<<<dim3((H4 * HH * 9 + 255) / 256), dim3(256), 0, s>>>(p.cw1, w.WT1, HH, H4, 9);
  k_conv_gemm<<<dim3(H4 / 16, T / 16, BB), wv, 0, s>>>(w.X1, w.WT1, p.cb1, w.SC, T, HH, 8, H4, 4, 9, 1);
  k_wconvt<<<dim3((HH * H4 * 9 + 255) / 256), dim3(256), 0, s>>>(p.cw2, w.WT2, H4, HH, 9);
  k_conv_gemm<<<dim3(HH / 16, T / 16, BB), wv, 0, s>>>(w.SC, w.WT2, p.cb2, w.TMP, T, H4, 10, HH, 4, 9, 0);
  k_add_ln<<<dim3(M), wv, 0, s>>>(w.X1, w.TMP, p.g2, p.b2, X);
}

static FFTP make_fftp(void* const* d_in, int base) {
  FFTP p;
  p.wqkv = (const float*)d_in[base + 0];  p.bqkv = (const float*)d_in[base + 1];
  p.wo   = (const float*)d_in[base + 2];  p.bo   = (const float*)d_in[base + 3];
  p.g1   = (const float*)d_in[base + 4];  p.b1   = (const float*)d_in[base + 5];
  p.cw1  = (const float*)d_in[base + 6];  p.cb1  = (const float*)d_in[base + 7];
  p.cw2  = (const float*)d_in[base + 8];  p.cb2  = (const float*)d_in[base + 9];
  p.g2   = (const float*)d_in[base + 10]; p.b2   = (const float*)d_in[base + 11];
  return p;
}

extern "C" void kernel_launch(void* const* d_in, const int* in_sizes, int n_in,
                              void* d_out, int out_size, void* d_ws, size_t ws_size,
                              hipStream_t stream) {
  (void)in_sizes; (void)n_in; (void)out_size; (void)ws_size;
  // ---- inputs (setup_inputs order; pytrees flattened in insertion order) ----
  const int*   ids     = (const int*)d_in[0];
  const int*   lengths = (const int*)d_in[1];
  const int*   durs    = (const int*)d_in[2];
  const float* featIn[5] = { (const float*)d_in[3], (const float*)d_in[4],
                             (const float*)d_in[5], (const float*)d_in[6],
                             (const float*)d_in[7] };        // pitch,energy,breath,rough,bright
  const float* emb   = (const float*)d_in[8];
  const float* mel_w = (const float*)d_in[187];
  const float* mel_b = (const float*)d_in[188];

  // ---- workspace layout ----
  char* ws = (char*)d_ws;
  size_t off = 0;
  auto alloc = [&](size_t bytes) { char* p = ws + off; off += (bytes + 255) & ~(size_t)255; return p; };
  const size_t SZX = (size_t)BB * TMEL * HH * sizeof(float);
  WsP w;
  w.X    = (float*)alloc(SZX);
  w.X1   = (float*)alloc(SZX);
  w.O    = (float*)alloc(SZX);
  w.TMP  = (float*)alloc(SZX);
  w.QKV  = (float*)alloc((size_t)BB * TMEL * H3 * 4);
  w.SC   = (float*)alloc((size_t)BB * NHH * TMEL * TMEL * 4);   // scores / FFN hidden
  w.XENC = (float*)alloc((size_t)BB * SS * HH * 4);
  w.TOK  = (int*)  alloc((size_t)BB * TMEL * 4);
  w.MLEN = (int*)  alloc(BB * 4);
  w.EF   = (float*)alloc((size_t)5 * BB * TMEL * 4);
  w.WT1  = (bf16*) alloc((size_t)H4 * HH * 9 * 2);
  w.WT2  = (bf16*) alloc((size_t)HH * H4 * 9 * 2);
  w.WQB  = (bf16*) alloc((size_t)H3 * HH * 2);
  w.WOB  = (bf16*) alloc((size_t)HH * HH * 2);
  w.WMEL = (bf16*) alloc((size_t)NMELC * HH * 2);
  w.VT   = (bf16*) alloc((size_t)BB * NHH * DH * TMEL * 2);

  // ---- output layout (concatenated, return order) ----
  float* out      = (float*)d_out;
  float* out_mel  = out;                                      // (B, 768, 80)
  const int PRED_N = BB * SS;
  float* out_pred = out + (size_t)BB * TMEL * NMELC;          // 6 x (B,S)
  float* out_mlen = out_pred + 6 * (size_t)PRED_N;            // (B,)

  // ---- 1) embedding ----
  k_embed<<<dim3(BB * SS), dim3(HH), 0, stream>>>(ids, emb, w.XENC);

  // ---- 2) encoder: 4 FFT blocks @ T=128 ----
  for (int i = 0; i < 4; ++i)
    run_fft_block(make_fftp(d_in, 9 + i * 12), w.XENC, SS, lengths, w, stream);

  // ---- 3) variance predictions: LN over singleton dim => clip(ob, lo, hi) ----
  const float plo[6] = { -4.f, -2.5f, -2.5f, 0.f,  0.f,  -2.5f };
  const float phi[6] = {  4.f,  2.5f,  2.5f, 0.8f, 1.5f,  2.5f };
  for (int j = 0; j < 6; ++j) {
    const float* obp = (const float*)d_in[105 + j * 12 + 11];
    k_fill_clip<<<dim3((PRED_N + 255) / 256), dim3(256), 0, stream>>>(
        obp, out_pred + (size_t)j * PRED_N, PRED_N, plo[j], phi[j]);
  }

  // ---- 4) length regulation ----
  k_lenreg<<<dim3(BB), dim3(128), 0, stream>>>(durs, w.TOK, w.MLEN, out_mlen);

  // ---- 5) expand + clip the five prosody features ----
  const float flo[5] = { -3.f, -3.f, 0.f, 0.f, -3.f };
  const float fhi[5] = {  3.f,  3.f, 1.f, 2.f,  3.f };
  const int EN = BB * TMEL;
  for (int f = 0; f < 5; ++f)
    k_expand<<<dim3((EN + 255) / 256), dim3(256), 0, stream>>>(
        featIn[f], w.TOK, w.EF + (size_t)f * EN, EN, flo[f], fhi[f]);

  // ---- 6) decoder input = regulated encoder states + variance-embedding convs ----
  VE5 ve;
  for (int f = 0; f < 5; ++f) {
    ve.w[f] = (const float*)d_in[177 + f * 2];
    ve.b[f] = (const float*)d_in[178 + f * 2];
  }
  k_build_dec<<<dim3(BB * TMEL), dim3(HH), 0, stream>>>(w.XENC, w.TOK, w.EF, ve, w.X);

  // ---- 7) decoder: 4 FFT blocks @ T=768, masked by mel_len ----
  for (int i = 0; i < 4; ++i)
    run_fft_block(make_fftp(d_in, 57 + i * 12), w.X, TMEL, w.MLEN, w, stream);

  // ---- 8) mel projection with clip(-10, 2), straight into d_out ----
  k_w2bf<<<dim3((NMELC * HH + 255) / 256), dim3(256), 0, stream>>>(mel_w, w.WMEL, NMELC * HH);
  k_linear<<<dim3(NMELC / 16, (BB * TMEL) / 16), dim3(32), 0, stream>>>(
      w.X, w.WMEL, mel_b, out_mel, BB * TMEL, NMELC, HH, 2, -10.f, 2.f);
}
